// Generator_59047210385625
// MI455X (gfx1250) — matrix operand order repaired and checked by reading, not compiled, not run on hardware
//
#include <hip/hip_runtime.h>
#include <math.h>

typedef __attribute__((ext_vector_type(16))) __bf16 v16bf;
typedef __attribute__((ext_vector_type(8)))  __bf16 v8bf;
typedef __attribute__((ext_vector_type(8)))  float  v8f;

#define B_   16
#define C_   192
#define TS   2048
#define TT   512
#define K2   384          // 2*C: [z | -0.5*z^2] x [m*s2r | s2r]
#define NEGV -1e9f

// ---------------- workspace layout (bytes) ----------------
#define SZ_ABUF  ((size_t)B_*TS*K2*2)          // 25,165,824  bf16 A  [b][t][k]
#define SZ_BBUF  ((size_t)B_*TT*K2*2)          //  6,291,456  bf16 B^T [b][u][k]
#define SZ_CONST ((size_t)B_*TT*4)             //     32,768  f32 per-(b,u) const
#define SZ_DIRS  ((size_t)B_*TS*32*2)          //  2,097,152  16 dir bits per lane per step
#define OFF_ABUF  ((size_t)0)
#define OFF_BBUF  (OFF_ABUF + SZ_ABUF)
#define OFF_CONST (OFF_BBUF + SZ_BBUF)
#define OFF_DIRS  (OFF_CONST + SZ_CONST)
#define OFF_JIDX  (OFF_DIRS + SZ_DIRS)

// ---------------- prep: A = [z | -0.5 z^2] transposed to [b][t][k] -------------
__global__ void prepA_kernel(const float* __restrict__ z_p, __bf16* __restrict__ Abuf) {
    size_t idx = (size_t)blockIdx.x * blockDim.x + threadIdx.x;   // B*TS*K2
    unsigned k  = (unsigned)(idx % K2);
    size_t   bt = idx / K2;
    unsigned t  = (unsigned)(bt & (TS - 1));
    unsigned b  = (unsigned)(bt >> 11);
    float v;
    if (k < C_) {
        v = z_p[((size_t)(b * C_ + k)) * TS + t];
    } else {
        float z = z_p[((size_t)(b * C_ + (k - C_))) * TS + t];
        v = -0.5f * z * z;
    }
    Abuf[idx] = (__bf16)v;
}

// ---------------- prep: B^T = [m*s2r | s2r] stored [b][u][k] -------------------
__global__ void prepB_kernel(const float* __restrict__ m_p,
                             const float* __restrict__ logs_p,
                             __bf16* __restrict__ Bbuf) {
    size_t idx = (size_t)blockIdx.x * blockDim.x + threadIdx.x;   // B*TT*K2
    unsigned k  = (unsigned)(idx % K2);
    size_t   bu = idx / K2;
    unsigned u  = (unsigned)(bu & (TT - 1));
    unsigned b  = (unsigned)(bu >> 9);
    unsigned c  = (k < C_) ? k : (k - C_);
    float s = __expf(-2.0f * logs_p[((size_t)(b * C_ + c)) * TT + u]);
    float v = (k < C_) ? m_p[((size_t)(b * C_ + c)) * TT + u] * s : s;
    Bbuf[idx] = (__bf16)v;
}

// ---------------- prep: const[b][u] = nc1 + nc4 --------------------------------
__global__ void prepConst_kernel(const float* __restrict__ m_p,
                                 const float* __restrict__ logs_p,
                                 float* __restrict__ cst) {
    int idx = blockIdx.x * blockDim.x + threadIdx.x;              // B*TT
    int u = idx & (TT - 1);
    int b = idx >> 9;
    float acc = 0.0f;
    for (int c = 0; c < C_; ++c) {
        float lg = logs_p[((size_t)(b * C_ + c)) * TT + u];
        float mm = m_p[((size_t)(b * C_ + c)) * TT + u];
        float s  = __expf(-2.0f * lg);
        acc += -0.91893853320467274f - lg - 0.5f * mm * mm * s;
    }
    cst[idx] = acc;
}

// ---------------- WMMA GEMM: 32x64 output block per wave -----------------------
// D = A(32x384) x B(384x64) + const, written into the path output region.
// Per K-chunk: 2 A-fragments reused across 4 B-fragments -> 8 WMMAs.
__global__ void negcent_gemm_kernel(const __bf16* __restrict__ Abuf,
                                    const __bf16* __restrict__ Bbuf,
                                    const float* __restrict__ cst,
                                    float* __restrict__ neg) {
    const int lane = threadIdx.x & 31;
    const int wg   = blockIdx.x * (blockDim.x >> 5) + (threadIdx.x >> 5);
    const int b    = wg >> 9;              // (TS/32)*(TT/64) = 512 blocks per batch
    const int rem  = wg & 511;
    const int tb   = rem >> 3;             // 0..63  : 32-row block
    const int ub   = rem & 7;              // 0..7   : 64-col block
    const int half = lane >> 4;            // ISA lane-group 0/1
    const int l15  = lane & 15;

    // A fragment (16-bit 16x32): lane row m=l15; element i -> K=base+half*16+i
    // (two contiguous 16B loads per tile-row)
    const __bf16* Arow0 = Abuf + ((size_t)(b * TS + tb * 32 + l15)) * K2 + half * 16;
    const __bf16* Arow1 = Arow0 + (size_t)16 * K2;
    // B fragment (16-bit 32x16): lane col n=l15; half i -> K=base+half*16+i
    // (one contiguous 32B load per tile-col from the transposed [u][k] store)
    const __bf16* Brow  = Bbuf + ((size_t)(b * TT + ub * 64 + l15)) * K2 + half * 16;

    v8f acc[2][4] = {};
#pragma unroll 2
    for (int kk = 0; kk < K2 / 32; ++kk) {
        union { v16bf v; v8bf h[2]; } a0, a1;
        a0.h[0] = *(const v8bf*)(Arow0 + kk * 32);
        a0.h[1] = *(const v8bf*)(Arow0 + kk * 32 + 8);
        a1.h[0] = *(const v8bf*)(Arow1 + kk * 32);
        a1.h[1] = *(const v8bf*)(Arow1 + kk * 32 + 8);
        v16bf bf[4];
#pragma unroll
        for (int n = 0; n < 4; ++n)
            bf[n] = *(const v16bf*)(Brow + (size_t)n * 16 * K2 + kk * 32);
#pragma unroll
        for (int n = 0; n < 4; ++n) {
            acc[0][n] = __builtin_amdgcn_wmma_f32_16x16x32_bf16(false, a0.v, false, bf[n],
                                                                (short)0, acc[0][n], false, false);
            acc[1][n] = __builtin_amdgcn_wmma_f32_16x16x32_bf16(false, a1.v, false, bf[n],
                                                                (short)0, acc[1][n], false, false);
        }
    }
#pragma unroll
    for (int n = 0; n < 4; ++n) {
        const int   col = ub * 64 + n * 16 + l15;
        const float cu  = cst[b * TT + col];
#pragma unroll
        for (int m = 0; m < 2; ++m) {
#pragma unroll
            for (int r = 0; r < 8; ++r) {
                int row = tb * 32 + m * 16 + r + half * 8;  // C/D layout
                neg[((size_t)(b * TS + row)) * TT + col] = acc[m][n][r] + cu;
            }
        }
    }
}

// ---------------- DP forward: ONE wave per batch, zero barriers -----------------
// Lane l owns j = l*16 .. l*16+15 in registers; halo via ds_bpermute (wave-sync).
// nc row loads are software-pipelined one t-step ahead.
__global__ void dp_forward_kernel(const float* __restrict__ neg,
                                  const int* __restrict__ spec_len,
                                  const int* __restrict__ text_len,
                                  unsigned short* __restrict__ dirs16) {
    const int b  = blockIdx.x;
    const int l  = threadIdx.x;            // 0..31
    const int sl = spec_len[b];
    const int tl = text_len[b];
    const int jbase = l * 16;

    float q[16];
#pragma unroll
    for (int i = 0; i < 16; ++i) q[i] = NEGV;
    if (l == 0) q[0] = 0.0f;

    const float4* p = (const float4*)(neg + (size_t)b * TS * TT + jbase); // stride 128 float4 per t
    unsigned short* db = dirs16 + (size_t)b * TS * 32 + l;

    union { float4 v[4]; float f[16]; } cur, nxt;
    cur.v[0] = p[0]; cur.v[1] = p[1]; cur.v[2] = p[2]; cur.v[3] = p[3];

    for (int t = 0; t < TS; ++t) {
        if (t + 1 < TS) {
            const float4* pn = p + (size_t)(t + 1) * (TT / 4);
            nxt.v[0] = pn[0]; nxt.v[1] = pn[1]; nxt.v[2] = pn[2]; nxt.v[3] = pn[3];
        }
        // halo = previous lane's old q[15]
        int   src  = (l == 0) ? 0 : (l - 1);
        float halo = __int_as_float(
            __builtin_amdgcn_ds_bpermute(src << 2, __float_as_int(q[15])));
        if (l == 0) halo = NEGV;

        const bool tvalid = (t < sl);
        unsigned d = 0;
        float qprev = halo;                 // old q[j-1]
#pragma unroll
        for (int i = 0; i < 16; ++i) {
            float qs = qprev;
            float qj = q[i];
            d |= (qs >= qj ? 1u : 0u) << i;
            bool  valid = tvalid && ((jbase + i) < tl);
            float ncv   = valid ? cur.f[i] : NEGV;
            qprev = qj;                     // save old q[i] before overwrite
            q[i]  = ncv + fmaxf(qj, qs);
        }
        db[(size_t)t * 32] = (unsigned short)d;
#pragma unroll
        for (int i = 0; i < 4; ++i) cur.v[i] = nxt.v[i];
    }
}

// ---------------- backtrack (dirs staged in LDS) + duration/logw/text_mask ------
__global__ void backtrack_kernel(const unsigned short* __restrict__ dirs16,
                                 const int* __restrict__ spec_len,
                                 const int* __restrict__ text_len,
                                 int* __restrict__ jidx,
                                 float* __restrict__ dur_out,
                                 float* __restrict__ logw_out,
                                 float* __restrict__ tmask_out) {
    extern __shared__ unsigned char smem[];          // TS*32 ushorts = 128 KB
    unsigned short* sd = (unsigned short*)smem;
    __shared__ int dur[TT];
    const int b = blockIdx.x;
    const int u = threadIdx.x;                       // 512 threads
    dur[u] = 0;

    // cooperative stage of this batch's direction bits into LDS (8192 uint4)
    const uint4* src = (const uint4*)(dirs16 + (size_t)b * TS * 32);
    uint4* dst = (uint4*)sd;
    for (int i = u; i < (TS * 32 * 2) / 16; i += TT) dst[i] = src[i];
    __syncthreads();

    if (u == 0) {
        const int sl = spec_len[b];
        const int tl = text_len[b];
        int* jb = jidx + b * TS;
        int j = 0;
        for (int t = TS - 1; t >= 0; --t) {
            int  j_eff = (t == sl - 1) ? (tl - 1) : j;
            bool valid = (t < sl);
            jb[t] = valid ? j_eff : -1;
            if (valid) {
                dur[j_eff] += 1;
                unsigned w = sd[t * 32 + (j_eff >> 4)];
                int came = (int)((w >> (j_eff & 15)) & 1u);
                j = j_eff - came;
                if (j < 0) j = 0;
            } else {
                j = j_eff;
            }
        }
    }
    __syncthreads();
    const int tl = text_len[b];
    float d  = (float)dur[u];
    float tm = (u < tl) ? 1.0f : 0.0f;
    dur_out[b * TT + u]   = d;
    logw_out[b * TT + u]  = logf(d + 1e-6f) * tm;
    tmask_out[b * TT + u] = tm;
}

// ---------------- one-hot path fill (overwrites neg_cent region) ---------------
__global__ void path_fill_kernel(const int* __restrict__ jidx, float* __restrict__ path) {
    size_t idx = (size_t)blockIdx.x * blockDim.x + threadIdx.x;   // B*TS*TT
    int u = (int)(idx & (TT - 1));
    int t = (int)((idx >> 9) & (TS - 1));
    int b = (int)(idx >> 20);
    path[idx] = (jidx[b * TS + t] == u) ? 1.0f : 0.0f;
}

// ---------------- m_p_exp / logs_p_exp gathers ---------------------------------
__global__ void gather_exp_kernel(const int* __restrict__ jidx,
                                  const float* __restrict__ m_p,
                                  const float* __restrict__ logs_p,
                                  float* __restrict__ m_exp,
                                  float* __restrict__ l_exp) {
    size_t idx = (size_t)blockIdx.x * blockDim.x + threadIdx.x;   // B*C*TS
    int t = (int)(idx & (TS - 1));
    int c = (int)((idx >> 11) % C_);
    int b = (int)(idx / ((size_t)C_ * TS));
    int j = jidx[b * TS + t];
    float mv = 0.0f, lv = 0.0f;
    if (j >= 0) {
        mv = m_p[((size_t)(b * C_ + c)) * TT + j];
        lv = logs_p[((size_t)(b * C_ + c)) * TT + j];
    }
    m_exp[idx] = mv;
    l_exp[idx] = lv;
}

// ---------------- spec_mask ----------------------------------------------------
__global__ void smask_kernel(const int* __restrict__ spec_len, float* __restrict__ out) {
    int idx = blockIdx.x * blockDim.x + threadIdx.x;              // B*TS
    int t = idx & (TS - 1);
    int b = idx >> 11;
    out[idx] = (t < spec_len[b]) ? 1.0f : 0.0f;
}

// ================================================================================
extern "C" void kernel_launch(void* const* d_in, const int* in_sizes, int n_in,
                              void* d_out, int out_size, void* d_ws, size_t ws_size,
                              hipStream_t stream) {
    const float* z_p      = (const float*)d_in[0];   // [16,192,2048]
    const float* m_p      = (const float*)d_in[1];   // [16,192,512]
    const float* logs_p   = (const float*)d_in[2];   // [16,192,512]
    const int*   spec_len = (const int*)d_in[3];     // [16]
    const int*   text_len = (const int*)d_in[4];     // [16]

    float* out   = (float*)d_out;
    float* path  = out;                                           // 16,777,216
    float* dur   = path  + (size_t)B_ * TS * TT;                  //      8,192
    float* logw  = dur   + (size_t)B_ * TT;                       //      8,192
    float* mexp  = logw  + (size_t)B_ * TT;                       //  6,291,456
    float* lexp  = mexp  + (size_t)B_ * C_ * TS;                  //  6,291,456
    float* tmask = lexp  + (size_t)B_ * C_ * TS;                  //      8,192
    float* smsk  = tmask + (size_t)B_ * TT;                       //     32,768

    char* ws = (char*)d_ws;
    __bf16*         Abuf   = (__bf16*)(ws + OFF_ABUF);
    __bf16*         Bbuf   = (__bf16*)(ws + OFF_BBUF);
    float*          cst    = (float*)(ws + OFF_CONST);
    unsigned short* dirs16 = (unsigned short*)(ws + OFF_DIRS);
    int*            jidx   = (int*)(ws + OFF_JIDX);

    // 1) build bf16 GEMM operands + additive constant
    prepA_kernel<<<(B_ * TS * K2) / 256, 256, 0, stream>>>(z_p, Abuf);
    prepB_kernel<<<(B_ * TT * K2) / 256, 256, 0, stream>>>(m_p, logs_p, Bbuf);
    prepConst_kernel<<<(B_ * TT) / 256, 256, 0, stream>>>(m_p, logs_p, cst);

    // 2) WMMA GEMM, 32x64 per wave: 8192 waves, 8 waves per 256-thread block
    negcent_gemm_kernel<<<(B_ * (TS / 32) * (TT / 64)) / 8, 256, 0, stream>>>(
        Abuf, Bbuf, cst, path);

    // 3) Viterbi forward: one wave32 per batch, register-resident q, no barriers
    dp_forward_kernel<<<B_, 32, 0, stream>>>(path, spec_len, text_len, dirs16);

    // 4) backtrack with LDS-staged dirs (128 KB dynamic LDS) + dur/logw/text_mask
    backtrack_kernel<<<B_, TT, TS * 32 * 2, stream>>>(dirs16, spec_len, text_len,
                                                      jidx, dur, logw, tmask);

    // 5) overwrite neg_cent region with the one-hot path
    path_fill_kernel<<<(int)(((size_t)B_ * TS * TT) / 256), 256, 0, stream>>>(jidx, path);

    // 6) expand prior stats along the path (gathers — path rows are one-hot)
    gather_exp_kernel<<<(int)(((size_t)B_ * C_ * TS) / 256), 256, 0, stream>>>(
        jidx, m_p, logs_p, mexp, lexp);

    // 7) spec_mask
    smask_kernel<<<(B_ * TS) / 256, 256, 0, stream>>>(spec_len, smsk);
}
